// MultiHeadAttn_59124519796956
// MI455X (gfx1250) — matrix operand order repaired and checked by reading, not compiled, not run on hardware
//
#include <hip/hip_runtime.h>
#include <hip/hip_bf16.h>

// ---------------------------------------------------------------------------
// MHA forward for MI455X (gfx1250, wave32, WMMA + async LDS staging).
// All GEMMs run in bf16 with f32 accumulation via v_wmma_f32_16x16x32_bf16.
// Tile staging uses GLOBAL_LOAD_ASYNC_TO_LDS_B128 (ASYNCcnt) where no
// transpose is needed, enabling a double-buffered GEMM pipeline.
// ---------------------------------------------------------------------------

typedef __attribute__((ext_vector_type(8)))  __bf16 v8bf;
typedef __attribute__((ext_vector_type(16))) __bf16 v16bf;
typedef __attribute__((ext_vector_type(8)))  float  v8f;

constexpr int kT  = 2048;
constexpr int kB  = 2;
constexpr int kD  = 1024;
constexpr int kH  = 16;
constexpr int kDH = 64;
constexpr int kM  = kT * kB;          // 4096 rows for all projections

// ---------------------------------------------------------------------------
// CDNA5 async memory helpers.
// global_load_async_to_lds_b128: per-lane 16B, memory -> LDS, ASYNCcnt.
// LDS address operand is the DS byte offset (low 32 bits of the generic ptr).
// ---------------------------------------------------------------------------
__device__ __forceinline__ void async_b128(void* lds_dst, const void* gsrc) {
  uint32_t ldsoff = (uint32_t)(uintptr_t)lds_dst;
  asm volatile("global_load_async_to_lds_b128 %0, %1, off"
               :
               : "v"(ldsoff), "v"(gsrc)
               : "memory");
}
__device__ __forceinline__ void wait_async0() {
  asm volatile("s_wait_asynccnt 0" ::: "memory");
}

// ---------------------------------------------------------------------------
// WMMA fragment helpers (layouts per CDNA5 ISA 7.12.2, wave32).
// A (16x32 bf16): lane L holds row M = L%16; elems 0..7 = K 8*half+0..7,
//                 elems 8..15 = K 8*half+16..23.
// B (32x16 bf16): lane L holds col N = L%16, same K order as A.
// C (16x16 f32):  VGPR r holds row M = r + 8*half, col N = L%16.
// ---------------------------------------------------------------------------
__device__ __forceinline__ v16bf frag_a_ld(const __bf16* row_base, int half) {
  v8bf lo = *reinterpret_cast<const v8bf*>(row_base + 8 * half);
  v8bf hi = *reinterpret_cast<const v8bf*>(row_base + 8 * half + 16);
  return __builtin_shufflevector(lo, hi, 0,1,2,3,4,5,6,7,8,9,10,11,12,13,14,15);
}

__device__ __forceinline__ v16bf frag_b_ld(const __bf16* row_base, int half) {
  v8bf lo = *reinterpret_cast<const v8bf*>(row_base + 8 * half);
  v8bf hi = *reinterpret_cast<const v8bf*>(row_base + 8 * half + 16);
  return __builtin_shufflevector(lo, hi, 0,1,2,3,4,5,6,7,8,9,10,11,12,13,14,15);
}

__device__ __forceinline__ v8f wmma_bf16(v16bf a, v16bf b, v8f c) {
  return __builtin_amdgcn_wmma_f32_16x16x32_bf16(
      false, a, false, b, (short)0, c, false, false);
}

// ---------------------------------------------------------------------------
// fp32 -> bf16 cast
// ---------------------------------------------------------------------------
__global__ __launch_bounds__(256)
void cast_f32_to_bf16(const float* __restrict__ src, __bf16* __restrict__ dst, int n) {
  int i = blockIdx.x * blockDim.x + threadIdx.x;
  int stride = gridDim.x * blockDim.x;
  for (; i < n; i += stride) dst[i] = (__bf16)src[i];
}

// ---------------------------------------------------------------------------
// Double-buffered bf16 GEMM: C(MxN) = A(MxK) * B(KxN); row-major.
// Block: 256 threads = 8 waves; block tile 128x128; K-tile 32.
// A tiles staged via async-to-LDS; B tiles staged through VGPRs (transpose)
// with global loads issued one tile ahead.
// ---------------------------------------------------------------------------
template <bool OUT_BF16>
__global__ __launch_bounds__(256)
void gemm_bf16_kernel(const __bf16* __restrict__ A, const __bf16* __restrict__ Bm,
                      void* __restrict__ Cp, int M, int N, int K) {
  constexpr int TK = 32;
  constexpr int PITCH = TK + 8;                 // 40 bf16 = 80 B (16B aligned)
  __shared__ __bf16 As[2][128][PITCH];          // [buf][m][k]
  __shared__ __bf16 Bs[2][128][PITCH];          // [buf][n][k] (transposed)

  const int tid  = threadIdx.x;
  const int lane = tid & 31;
  const int wave = tid >> 5;
  const int wm   = wave & 3;                    // 0..3 -> 32-row strips
  const int wn   = wave >> 2;                   // 0..1 -> 64-col strips
  const int half = lane >> 4;
  const int l16  = lane & 15;
  const int m0   = blockIdx.x * 128;
  const int n0   = blockIdx.y * 128;

  // staging coordinates
  const int arow = tid >> 1, ac0 = (tid & 1) * 16;   // A: 2 thr/row, 16 bf16
  const int bkk  = tid >> 3, bnb = (tid & 7) * 16;   // B: 8 thr/k-row, 16 bf16

  auto stageA = [&](int k0, int pb) {
    const char* g = (const char*)(A + (size_t)(m0 + arow) * K + k0 + ac0);
    async_b128(&As[pb][arow][ac0],     g);
    async_b128(&As[pb][arow][ac0 + 8], g + 16);
  };
  auto loadB = [&](int k0, v8bf br[2]) {
    const v8bf* g = reinterpret_cast<const v8bf*>(Bm + (size_t)(k0 + bkk) * N + n0 + bnb);
    br[0] = g[0];
    br[1] = g[1];
  };
  auto storeB = [&](int pb, v8bf br[2]) {
#pragma unroll
    for (int i = 0; i < 8; ++i) Bs[pb][bnb + i][bkk]     = br[0][i];
#pragma unroll
    for (int i = 0; i < 8; ++i) Bs[pb][bnb + 8 + i][bkk] = br[1][i];
  };

  v8f acc[2][4] = {};

  const int nt = K / TK;
  // prologue: stage tile 0 into buffer 0
  stageA(0, 0);
  {
    v8bf br[2];
    loadB(0, br);
    storeB(0, br);
  }
  wait_async0();
  __syncthreads();

  int p = 0;
  for (int t = 0; t < nt; ++t) {
    const bool nxt = (t + 1 < nt);
    v8bf br[2];
    if (nxt) {
      stageA((t + 1) * TK, p ^ 1);              // async, overlaps compute
      loadB((t + 1) * TK, br);                  // global loads in flight
    }
    if (t + 2 < nt) {
      __builtin_prefetch(Bm + (size_t)((t + 2) * TK + bkk) * N + n0 + bnb, 0, 1);
    }

    // --- compute tile t from buffer p ---
    v16bf af[2], bfrg[4];
#pragma unroll
    for (int i = 0; i < 2; ++i)
      af[i] = frag_a_ld(&As[p][wm * 32 + i * 16 + l16][0], half);
#pragma unroll
    for (int j = 0; j < 4; ++j)
      bfrg[j] = frag_b_ld(&Bs[p][wn * 64 + j * 16 + l16][0], half);
#pragma unroll
    for (int i = 0; i < 2; ++i)
#pragma unroll
      for (int j = 0; j < 4; ++j)
        acc[i][j] = wmma_bf16(af[i], bfrg[j], acc[i][j]);

    if (nxt) {
      storeB(p ^ 1, br);                        // transpose scatter after compute
      wait_async0();                            // A tile t+1 fully in LDS
    }
    __syncthreads();
    p ^= 1;
  }

  // --- epilogue: C layout -> global ---
#pragma unroll
  for (int i = 0; i < 2; ++i) {
#pragma unroll
    for (int j = 0; j < 4; ++j) {
#pragma unroll
      for (int r = 0; r < 8; ++r) {
        int row = m0 + wm * 32 + i * 16 + half * 8 + r;
        int col = n0 + wn * 64 + j * 16 + l16;
        float v = acc[i][j][r];
        if (OUT_BF16) ((__bf16*)Cp)[(size_t)row * N + col] = (__bf16)v;
        else          ((float*) Cp)[(size_t)row * N + col] = v;
      }
    }
  }
}

// ---------------------------------------------------------------------------
// Flash attention (causal). Grid: (T/64, B*H). Block: 128 threads = 4 waves.
// Q/K tiles staged async-to-LDS; V transposed through VGPRs.
// Online softmax on 16-lane groups (C-layout row residency) via shfl_xor.
// ---------------------------------------------------------------------------
__global__ __launch_bounds__(128)
void flash_attn_kernel(const __bf16* __restrict__ Q, const __bf16* __restrict__ KV,
                       __bf16* __restrict__ O) {
  constexpr int PITCH = 72;                     // 64 + 8 pad
  __shared__ __bf16 Qs[64][PITCH];              // [m][d]
  __shared__ __bf16 Ks[64][PITCH];              // [j][d]  (B operand of Q*K^T)
  __shared__ __bf16 Vs[64][PITCH];              // [d][j]  (B operand of P*V)
  __shared__ __bf16 Ps[4][16][PITCH];           // per-wave P strip [m][j]

  const int tid  = threadIdx.x;
  const int lane = tid & 31;
  const int wave = tid >> 5;
  const int half = lane >> 4;
  const int l16  = lane & 15;
  const int i0   = blockIdx.x * 64;
  const int bh   = blockIdx.y;
  const int b    = bh >> 4;
  const int head = bh & 15;
  const float scale = 0.125f;                   // 1/sqrt(DH)

  const int srow = tid >> 1, sc0 = (tid & 1) * 32;   // 2 thr/row, 32 bf16 each

  // --- stage Q tile async (waited before first-iteration barrier) ---
  {
    const char* g = (const char*)(Q + (size_t)((i0 + srow) * kB + b) * (kH * kDH) +
                                  head * kDH + sc0);
#pragma unroll
    for (int i = 0; i < 4; ++i)
      async_b128(&Qs[srow][sc0 + 8 * i], g + 16 * i);
  }

  v8f o[4] = {};
  float mrun[8], lrun[8];
#pragma unroll
  for (int r = 0; r < 8; ++r) { mrun[r] = -1e30f; lrun[r] = 0.f; }

  for (int j0 = 0; j0 <= i0; j0 += 64) {
    // --- stage K tile [j][d] async ---
    {
      const char* g = (const char*)(KV + (size_t)((j0 + srow) * kB + b) * (2 * kH * kDH) +
                                    head * 2 * kDH + sc0);
#pragma unroll
      for (int i = 0; i < 4; ++i)
        async_b128(&Ks[srow][sc0 + 8 * i], g + 16 * i);
    }
    // --- stage V tile transposed [d][j] through VGPRs ---
    {
      const v8bf* g = reinterpret_cast<const v8bf*>(
          KV + (size_t)((j0 + srow) * kB + b) * (2 * kH * kDH) +
          head * 2 * kDH + kDH + sc0);
      v8bf vr[4];
#pragma unroll
      for (int i = 0; i < 4; ++i) vr[i] = g[i];
#pragma unroll
      for (int i = 0; i < 4; ++i)
#pragma unroll
        for (int e = 0; e < 8; ++e)
          Vs[sc0 + i * 8 + e][srow] = vr[i][e];
    }
    wait_async0();
    __syncthreads();

    // --- S = Q * K^T (two 32-wide K slices over DH=64) ---
    v8f s[4] = {};
#pragma unroll
    for (int ks = 0; ks < 2; ++ks) {
      v16bf qa = frag_a_ld(&Qs[wave * 16 + l16][ks * 32], half);
#pragma unroll
      for (int nf = 0; nf < 4; ++nf) {
        v16bf kb = frag_b_ld(&Ks[nf * 16 + l16][ks * 32], half);
        s[nf] = wmma_bf16(qa, kb, s[nf]);
      }
    }

    // --- scale + causal mask ---
    const int rowbase = i0 + wave * 16 + half * 8;
#pragma unroll
    for (int nf = 0; nf < 4; ++nf) {
#pragma unroll
      for (int r = 0; r < 8; ++r) {
        int gi = rowbase + r;
        int gj = j0 + nf * 16 + l16;
        float v = s[nf][r] * scale;
        s[nf][r] = (gj > gi) ? -1e30f : v;
      }
    }

    // --- online softmax: row max ---
    float mnew[8];
#pragma unroll
    for (int r = 0; r < 8; ++r) {
      float m = s[0][r];
#pragma unroll
      for (int nf = 1; nf < 4; ++nf) m = fmaxf(m, s[nf][r]);
      mnew[r] = m;
    }
#pragma unroll
    for (int d = 1; d < 16; d <<= 1)
#pragma unroll
      for (int r = 0; r < 8; ++r) mnew[r] = fmaxf(mnew[r], __shfl_xor(mnew[r], d, 32));

    float alpha[8];
#pragma unroll
    for (int r = 0; r < 8; ++r) {
      float mi = fmaxf(mrun[r], mnew[r]);
      alpha[r] = __expf(mrun[r] - mi);
      mrun[r]  = mi;
    }

    // --- P = exp(S - m), row sums ---
    float lpart[8];
#pragma unroll
    for (int r = 0; r < 8; ++r) lpart[r] = 0.f;
#pragma unroll
    for (int nf = 0; nf < 4; ++nf) {
#pragma unroll
      for (int r = 0; r < 8; ++r) {
        float p = __expf(s[nf][r] - mrun[r]);
        s[nf][r] = p;
        lpart[r] += p;
      }
    }
#pragma unroll
    for (int d = 1; d < 16; d <<= 1)
#pragma unroll
      for (int r = 0; r < 8; ++r) lpart[r] += __shfl_xor(lpart[r], d, 32);
#pragma unroll
    for (int r = 0; r < 8; ++r) lrun[r] = lrun[r] * alpha[r] + lpart[r];

    // --- rescale O, round-trip P through LDS (C-layout -> A-layout) ---
#pragma unroll
    for (int nf = 0; nf < 4; ++nf)
#pragma unroll
      for (int r = 0; r < 8; ++r) o[nf][r] *= alpha[r];

#pragma unroll
    for (int nf = 0; nf < 4; ++nf)
#pragma unroll
      for (int r = 0; r < 8; ++r)
        Ps[wave][half * 8 + r][nf * 16 + l16] = (__bf16)s[nf][r];
    // same-wave LDS ops are in order: store->load RAW is safe without barrier

    // --- O += P * V ---
#pragma unroll
    for (int ks = 0; ks < 2; ++ks) {
      v16bf pa = frag_a_ld(&Ps[wave][l16][ks * 32], half);
#pragma unroll
      for (int nf = 0; nf < 4; ++nf) {
        v16bf vb = frag_b_ld(&Vs[nf * 16 + l16][ks * 32], half);
        o[nf] = wmma_bf16(pa, vb, o[nf]);
      }
    }
    __syncthreads();
  }

  // --- epilogue: O /= l, store bf16 ---
  float inv[8];
#pragma unroll
  for (int r = 0; r < 8; ++r) inv[r] = 1.0f / lrun[r];
#pragma unroll
  for (int nf = 0; nf < 4; ++nf) {
#pragma unroll
    for (int r = 0; r < 8; ++r) {
      int t = i0 + wave * 16 + half * 8 + r;
      O[(size_t)(t * kB + b) * (kH * kDH) + head * kDH + nf * 16 + l16] =
          (__bf16)(o[nf][r] * inv[r]);
    }
  }
}

// ---------------------------------------------------------------------------
// Residual + LayerNorm (fp32). One block per row, 256 threads x 4 elems.
// ---------------------------------------------------------------------------
__global__ __launch_bounds__(256)
void residual_ln_kernel(const float* __restrict__ h, const float* __restrict__ proj,
                        const float* __restrict__ gamma, const float* __restrict__ beta,
                        float* __restrict__ out) {
  const int m    = blockIdx.x;
  const int tid  = threadIdx.x;
  const int lane = tid & 31;
  const int wave = tid >> 5;
  __shared__ float rS[8], rS2[8];

  float x[4], s = 0.f, s2 = 0.f;
#pragma unroll
  for (int e = 0; e < 4; ++e) {
    int c = tid + e * 256;
    x[e] = h[(size_t)m * kD + c] + proj[(size_t)m * kD + c];
    s += x[e];
    s2 += x[e] * x[e];
  }
#pragma unroll
  for (int d = 1; d < 32; d <<= 1) {
    s  += __shfl_xor(s, d, 32);
    s2 += __shfl_xor(s2, d, 32);
  }
  if (lane == 0) { rS[wave] = s; rS2[wave] = s2; }
  __syncthreads();
  s = 0.f; s2 = 0.f;
#pragma unroll
  for (int i = 0; i < 8; ++i) { s += rS[i]; s2 += rS2[i]; }

  const float mu  = s * (1.0f / kD);
  const float var = s2 * (1.0f / kD) - mu * mu;
  const float rst = rsqrtf(var + 1e-5f);
#pragma unroll
  for (int e = 0; e < 4; ++e) {
    int c = tid + e * 256;
    out[(size_t)m * kD + c] = (x[e] - mu) * rst * gamma[c] + beta[c];
  }
}

// ---------------------------------------------------------------------------
// Launch: cast -> QKV projections (WMMA) -> flash attention (WMMA)
//         -> output projection (WMMA) -> residual+LN.
// ---------------------------------------------------------------------------
extern "C" void kernel_launch(void* const* d_in, const int* in_sizes, int n_in,
                              void* d_out, int out_size, void* d_ws, size_t ws_size,
                              hipStream_t stream) {
  const float* h     = (const float*)d_in[0];
  // d_in[1] = attn_mask (bool, causal) -- computed analytically in-kernel
  const float* Wq    = (const float*)d_in[2];
  const float* Wkv   = (const float*)d_in[3];
  const float* Wo    = (const float*)d_in[4];
  const float* gamma = (const float*)d_in[5];
  const float* beta  = (const float*)d_in[6];
  float* out = (float*)d_out;

  char* ws = (char*)d_ws;
  size_t off = 0;
  auto alloc = [&](size_t bytes) -> void* {
    off = (off + 255) & ~(size_t)255;
    void* p = ws + off;
    off += bytes;
    return p;
  };

  __bf16* hbf   = (__bf16*)alloc((size_t)kM * kD * 2);          // 8 MB
  __bf16* Wqbf  = (__bf16*)alloc((size_t)kD * kD * 2);          // 2 MB
  __bf16* Wkvbf = (__bf16*)alloc((size_t)kD * 2 * kD * 2);      // 4 MB
  __bf16* Wobf  = (__bf16*)alloc((size_t)kD * kD * 2);          // 2 MB
  __bf16* Qbf   = (__bf16*)alloc((size_t)kM * kD * 2);          // 8 MB
  __bf16* KVbf  = (__bf16*)alloc((size_t)kM * 2 * kD * 2);      // 16 MB
  __bf16* Obf   = (__bf16*)alloc((size_t)kM * kD * 2);          // 8 MB
  float*  proj  = (float*) alloc((size_t)kM * kD * 4);          // 16 MB

  // 1) casts
  cast_f32_to_bf16<<<1024, 256, 0, stream>>>(h,   hbf,   kM * kD);
  cast_f32_to_bf16<<<1024, 256, 0, stream>>>(Wq,  Wqbf,  kD * kD);
  cast_f32_to_bf16<<<1024, 256, 0, stream>>>(Wkv, Wkvbf, kD * 2 * kD);
  cast_f32_to_bf16<<<1024, 256, 0, stream>>>(Wo,  Wobf,  kD * kD);

  // 2) projections: Q = h*Wq (4096x1024x1024), KV = h*Wkv (4096x2048x1024)
  gemm_bf16_kernel<true><<<dim3(kM / 128, kD / 128), 256, 0, stream>>>(
      hbf, Wqbf, Qbf, kM, kD, kD);
  gemm_bf16_kernel<true><<<dim3(kM / 128, (2 * kD) / 128), 256, 0, stream>>>(
      hbf, Wkvbf, KVbf, kM, 2 * kD, kD);

  // 3) flash attention: grid (T/64, B*H)
  flash_attn_kernel<<<dim3(kT / 64, kB * kH), 128, 0, stream>>>(Qbf, KVbf, Obf);

  // 4) output projection: proj = O * Wo (f32 out)
  gemm_bf16_kernel<false><<<dim3(kM / 128, kD / 128), 256, 0, stream>>>(
      Obf, Wobf, proj, kM, kD, kD);

  // 5) residual + layernorm
  residual_ln_kernel<<<kM, 256, 0, stream>>>(h, proj, gamma, beta, out);
}
